// Bn_bin_conv_pool_block_baw_Threshold_27101243638458
// MI455X (gfx1250) — 8-bit matrix operand order repaired and checked by reading, compiled, not run on hardware
//
#include <hip/hip_runtime.h>

typedef int v8i __attribute__((ext_vector_type(8)));
typedef int v4i __attribute__((ext_vector_type(4)));

#define NB      64
#define CIN     64
#define LIN     16384
#define COUT    128
#define KW      7
#define PADLO   3
#define LP      16424
#define LOUT    8189
#define NGRP    64
#define NTILE   17
#define LDS_STR 276
#define LDS_BYTES (COUT * LDS_STR * 4)

__global__ void binarize_input(const float* __restrict__ I,
                               signed char* __restrict__ bt) {
    int idx = blockIdx.x * blockDim.x + threadIdx.x;
    const int total = NB * LP * CIN;
    if (idx >= total) return;
    int ci = idx & (CIN - 1);
    int p  = (idx / CIN) % LP;
    int b  = idx / (CIN * LP);
    signed char v = -1;
    int l = p - PADLO;
    if (l >= 0 && l < LIN) {
        float x = I[((long long)b * CIN + ci) * LIN + l];
        v = (x >= 0.0f) ? (signed char)1 : (signed char)-1;
    }
    bt[idx] = v;
}

__global__ void pack_weights(const float* __restrict__ W,
                             int* __restrict__ awp) {
    int t = blockIdx.x * blockDim.x + threadIdx.x;
    if (t >= KW * 8 * 32) return;
    int lane = t & 31;
    int cot  = (t >> 5) & 7;
    int k    = t >> 8;
    int m    = cot * 16 + (lane & 15);
    int half = lane >> 4;
#pragma unroll
    for (int v = 0; v < 8; ++v) {
        int kb = (v >> 2) * 32 + half * 16 + (v & 3) * 4;
        unsigned int d = 0;
#pragma unroll
        for (int j = 0; j < 4; ++j) {
            float w = W[(m * CIN + (kb + j)) * KW + k];
            signed char s = (w >= 0.0f) ? (signed char)1 : (signed char)-1;
            d |= ((unsigned int)(unsigned char)s) << (8 * j);
        }
        awp[t * 8 + v] = (int)d;
    }
}

__global__ void __launch_bounds__(256)
conv_pool_threshold(const signed char* __restrict__ bt,
                    const int* __restrict__ awp,
                    const float* __restrict__ tp,
                    const float* __restrict__ tm,
                    const float* __restrict__ ps,
                    const float* __restrict__ ms,
                    float* __restrict__ out) {
    extern __shared__ int conv_lds[];

    const int G      = blockIdx.x & (NGRP - 1);
    const int b      = blockIdx.x / NGRP;
    const int tid    = threadIdx.x;
    const int wave   = tid >> 5;
    const int lane   = tid & 31;
    const int n      = lane & 15;
    const int half   = lane >> 4;
    const int parity = wave & 1;
    const int cot0   = (wave >> 1) * 2;
    const int c0     = G * 256;

    const signed char* btb = bt + (long long)b * LP * CIN;

    v8i af[2][KW];
#pragma unroll
    for (int c2 = 0; c2 < 2; ++c2) {
#pragma unroll
        for (int k = 0; k < KW; ++k) {
            const int* ap = awp + ((k * 8 + (cot0 + c2)) * 32 + lane) * 8;
            v4i alo = *(const v4i*)(ap);
            v4i ahi = *(const v4i*)(ap + 4);
            af[c2][k][0] = alo[0]; af[c2][k][1] = alo[1];
            af[c2][k][2] = alo[2]; af[c2][k][3] = alo[3];
            af[c2][k][4] = ahi[0]; af[c2][k][5] = ahi[1];
            af[c2][k][6] = ahi[2]; af[c2][k][7] = ahi[3];
        }
    }

    for (int t = parity; t < NTILE; t += 2) {
        const int pos = c0 + t * 16 + n;

        v8i bf[KW];
#pragma unroll
        for (int k = 0; k < KW; ++k) {
            const signed char* bp = btb + (long long)(pos + k) * CIN + half * 16;
            v4i blo = *(const v4i*)(bp);
            v4i bhi = *(const v4i*)(bp + 32);
            bf[k][0] = blo[0]; bf[k][1] = blo[1]; bf[k][2] = blo[2]; bf[k][3] = blo[3];
            bf[k][4] = bhi[0]; bf[k][5] = bhi[1]; bf[k][6] = bhi[2]; bf[k][7] = bhi[3];
        }

        v8i acc0 = {0, 0, 0, 0, 0, 0, 0, 0};
        v8i acc1 = {0, 0, 0, 0, 0, 0, 0, 0};
#pragma unroll
        for (int k = 0; k < KW; ++k) {
            acc0 = __builtin_amdgcn_wmma_i32_16x16x64_iu8(
                true, af[0][k], true, bf[k], acc0, false, false);
            acc1 = __builtin_amdgcn_wmma_i32_16x16x64_iu8(
                true, af[1][k], true, bf[k], acc1, false, false);
        }

#pragma unroll
        for (int v = 0; v < 8; ++v) {
            int co0 = cot0 * 16 + v + 8 * half;
            conv_lds[co0 * LDS_STR + t * 16 + n] = acc0[v];
            conv_lds[(co0 + 16) * LDS_STR + t * 16 + n] = acc1[v];
        }
    }
    __syncthreads();

    for (int idx = tid; idx < COUT * 128; idx += 256) {
        int co = idx >> 7;
        int ol = idx & 127;
        int o  = G * 128 + ol;
        if (o >= LOUT) continue;
        const int* row = conv_lds + co * LDS_STR + 2 * ol;
        int m = row[0];
#pragma unroll
        for (int j = 1; j < 7; ++j) m = (row[j] > m) ? row[j] : m;
        float pooled = (float)m;
        float r;
        if (pooled >= 0.0f) r = (pooled > tp[co]) ? ps[co] : -ps[co];
        else                r = (pooled > tm[co]) ? ms[co] : -ms[co];
        out[((long long)b * COUT + co) * LOUT + o] = r;
    }
}

extern "C" void kernel_launch(void* const* d_in, const int* in_sizes, int n_in,
                              void* d_out, int out_size, void* d_ws, size_t ws_size,
                              hipStream_t stream) {
    const float* I  = (const float*)d_in[0];
    const float* W  = (const float*)d_in[1];
    const float* tp = (const float*)d_in[2];
    const float* tm = (const float*)d_in[3];
    const float* ps = (const float*)d_in[4];
    const float* ms = (const float*)d_in[5];
    float* out = (float*)d_out;

    int*         awp = (int*)d_ws;
    signed char* bt  = (signed char*)d_ws + 65536;

    const int nbin = NB * LP * CIN;
    binarize_input<<<(nbin + 255) / 256, 256, 0, stream>>>(I, bt);
    pack_weights<<<(KW * 8 * 32 + 255) / 256, 256, 0, stream>>>(W, awp);
    conv_pool_threshold<<<NB * NGRP, 256, LDS_BYTES, stream>>>(
        bt, awp, tp, tm, ps, ms, out);
}
